// entropy_33818572488707
// MI455X (gfx1250) — compile-verified, matrix operand order checked by reading, not run on hardware
//
#include <hip/hip_runtime.h>
#include <math.h>

// ---------------- types ----------------
typedef __bf16    v16bf __attribute__((ext_vector_type(16)));
typedef float     v8f   __attribute__((ext_vector_type(8)));
typedef unsigned  v4u   __attribute__((ext_vector_type(4)));
typedef int       v8i   __attribute__((ext_vector_type(8)));
typedef int       v4i   __attribute__((ext_vector_type(4)));

#define KDIM 768
#define BM   128
#define BN   128
#define BK   64            // K-slab per TDM transfer (two WMMA k-chunks)
#define LDSS 72            // LDS row stride in bf16 elems (128B data + 16B pad)
#define NK   (KDIM / BK)   // 12

// 0.5*log(2/pi)
#define LOGC (-0.22579135264472741f)

// ---------------- helpers ----------------
__device__ __forceinline__ unsigned short f2bf(float x) {
    unsigned u = __float_as_uint(x);
    unsigned r = u + 0x7fffu + ((u >> 16) & 1u);  // round-to-nearest-even
    return (unsigned short)(r >> 16);
}
__device__ __forceinline__ unsigned pack2bf(float a, float b) {
    return (unsigned)f2bf(a) | ((unsigned)f2bf(b) << 16);
}
// monotone float <-> uint mapping so atomicMax(unsigned) == float max
__device__ __forceinline__ unsigned f2ord(float f) {
    unsigned u = __float_as_uint(f);
    return (u & 0x80000000u) ? ~u : (u | 0x80000000u);
}
__device__ __forceinline__ float ord2f(unsigned u) {
    return __uint_as_float((u & 0x80000000u) ? (u & 0x7fffffffu) : ~u);
}

// ---------------- TDM tile load: 128 rows x 64 bf16, row-padded into LDS ----------------
// D# group0: count=1, lds_addr, 57-bit global addr, type=2
// D# group1: data_size=2B, pad_enable, pad_interval=32 DWORDs (one 128B row),
//            pad_amount=4 DWORDs (16B), tensor_dim0=KDIM, tensor_dim1=nrows,
//            tile_dim0=BK, tile_dim1=BM, tensor_dim0_stride=KDIM
__device__ __forceinline__ void tdm_load_tile(const unsigned short* gbase,
                                              int row0, int k0, int nrows,
                                              unsigned lds_addr) {
    unsigned long long ga = (unsigned long long)(uintptr_t)gbase
                          + ((unsigned long long)row0 * KDIM + (unsigned)k0) * 2ull;
    v4u g0;
    g0.x = 1u;                                           // count=1, user mode
    g0.y = lds_addr;                                     // LDS byte address
    g0.z = (unsigned)(ga & 0xffffffffull);               // global_addr[31:0]
    g0.w = (unsigned)((ga >> 32) & 0x01ffffffull)        // global_addr[56:32]
         | (2u << 30);                                   // type=2 (image)
    v8i g1;
    g1[0] = (int)((1u << 16)      // data_size = 2 bytes
                | (1u << 20)      // pad_enable
                | (4u << 22)      // pad_interval: 32 DWORDs (one 128B row)
                | (3u << 25));    // pad_amount: 4 DWORDs (16B)
    g1[1] = (int)(((unsigned)KDIM & 0xffffu) << 16);                 // tensor_dim0[15:0]
    g1[2] = (int)(((unsigned)KDIM >> 16) | (((unsigned)nrows & 0xffffu) << 16)); // dim0 hi | dim1 lo
    g1[3] = (int)(((unsigned)nrows >> 16) | ((unsigned)BK << 16));   // dim1 hi | tile_dim0
    g1[4] = BM;      // tile_dim1 = 128 rows (tile_dim2 = 0)
    g1[5] = KDIM;    // tensor_dim0_stride[31:0]
    g1[6] = 0;
    g1[7] = 0;
    v4i z4 = {0, 0, 0, 0};
#if __has_include(<hip/amd_detail/amd_gfx1250_TDM.h>)
    v8i z8 = {0, 0, 0, 0, 0, 0, 0, 0};
    __builtin_amdgcn_tensor_load_to_lds(g0, g1, z4, z4, z8, 0);
#else
    __builtin_amdgcn_tensor_load_to_lds(g0, g1, z4, z4, 0);
#endif
}

// ---------------- kernel 1: fused norms + scale + bf16 convert ----------------
// one wave per row; row lives in 24 VGPRs/lane; writes ex/ey scaled by 1/norm as bf16
__global__ void convert_kernel(const float* __restrict__ ex,
                               const float* __restrict__ ey,
                               unsigned short* __restrict__ exb,
                               unsigned short* __restrict__ eyb,
                               int Nx, int Ny) {
    int gw   = (blockIdx.x * blockDim.x + threadIdx.x) >> 5;
    int lane = threadIdx.x & 31;
    if (gw >= Nx + Ny) return;
    const float*    src = (gw < Nx) ? (ex  + (size_t)gw * KDIM) : (ey  + (size_t)(gw - Nx) * KDIM);
    unsigned short* dst = (gw < Nx) ? (exb + (size_t)gw * KDIM) : (eyb + (size_t)(gw - Nx) * KDIM);
    float4 f[6];
    float  s = 0.f;
    const float4* src4 = (const float4*)src;   // 192 float4 per row
#pragma unroll
    for (int i = 0; i < 6; ++i) {
        f[i] = src4[lane + i * 32];
        s += f[i].x * f[i].x + f[i].y * f[i].y + f[i].z * f[i].z + f[i].w * f[i].w;
    }
#pragma unroll
    for (int o = 16; o > 0; o >>= 1) s += __shfl_xor(s, o, 32);   // all lanes get sum
    float r = 1.0f / fmaxf(sqrtf(s), 1e-8f);
    uint2* dst2 = (uint2*)dst;
#pragma unroll
    for (int i = 0; i < 6; ++i) {
        uint2 p;
        p.x = pack2bf(f[i].x * r, f[i].y * r);
        p.y = pack2bf(f[i].z * r, f[i].w * r);
        dst2[lane + i * 32] = p;
    }
}

// ---------------- kernel 2: init row-max buffer ----------------
__global__ void init_kernel(unsigned* __restrict__ rowmaxU, int n) {
    int i = blockIdx.x * blockDim.x + threadIdx.x;
    if (i < n) rowmaxU[i] = 0u;
}

// ---------------- shared compute step: fragments + 2x4 WMMA for one K=32 chunk ----------
__device__ __forceinline__ void wmma_step(const unsigned short* As,
                                          const unsigned short* Bs,
                                          int wm, int wn, int m, int kbase,
                                          v8f acc[2][4]) {
    union FragU { uint4 q[2]; v16bf v; };
    FragU fb[4];
#pragma unroll
    for (int nt = 0; nt < 4; ++nt) {
        int r = wn * 64 + nt * 16 + m;
        fb[nt].q[0] = *(const uint4*)&Bs[r * LDSS + kbase];
        fb[nt].q[1] = *(const uint4*)&Bs[r * LDSS + kbase + 16];
    }
#pragma unroll
    for (int mt = 0; mt < 2; ++mt) {
        FragU fa;
        int r = wm * 32 + mt * 16 + m;
        fa.q[0] = *(const uint4*)&As[r * LDSS + kbase];
        fa.q[1] = *(const uint4*)&As[r * LDSS + kbase + 16];
#pragma unroll
        for (int nt = 0; nt < 4; ++nt) {
            acc[mt][nt] = __builtin_amdgcn_wmma_f32_16x16x32_bf16(
                false, fa.v, false, fb[nt].v, (short)0, acc[mt][nt], false, false);
        }
    }
}

// ---------------- shared epilogue: row max + atomic combine ----------------
__device__ __forceinline__ void rowmax_epilogue(v8f acc[2][4], unsigned* rowmaxU,
                                                int rowBase, int wm, int lane, int hi) {
#pragma unroll
    for (int mt = 0; mt < 2; ++mt) {
#pragma unroll
        for (int r = 0; r < 8; ++r) {
            float v = acc[mt][0][r];
            v = fmaxf(v, acc[mt][1][r]);
            v = fmaxf(v, acc[mt][2][r]);
            v = fmaxf(v, acc[mt][3][r]);
#pragma unroll
            for (int o = 1; o < 16; o <<= 1) v = fmaxf(v, __shfl_xor(v, o, 16));
            if ((lane & 15) == 0) {
                int row = rowBase + wm * 32 + mt * 16 + hi * 8 + r;
                atomicMax(&rowmaxU[row], f2ord(v));
            }
        }
    }
}

// ---------------- kernel 3a: TDM double-buffered bf16 WMMA GEMM + row max ----------------
__global__ __launch_bounds__(256)
void gemm_tdm_kernel(const unsigned short* __restrict__ exb,
                     const unsigned short* __restrict__ eyb,
                     unsigned* __restrict__ rowmaxU, int Nx, int Ny) {
    __shared__ unsigned short As[2][BM * LDSS];
    __shared__ unsigned short Bs[2][BN * LDSS];

    const int t    = threadIdx.x;
    const int lane = t & 31;
    const int wave = t >> 5;          // 0..7
    const int wm   = wave >> 1;       // 0..3 : 32-row band
    const int wn   = wave & 1;        // 0..1 : 64-col band
    const int m    = lane & 15;
    const int hi   = lane >> 4;
    const int kA0  = hi * 8;          // bf16 offset of this half-wave's first K chunk

    const int rowBase = blockIdx.y * BM;
    const int colBase = blockIdx.x * BN;

    const unsigned ldsA0 = (unsigned)(uintptr_t)&As[0][0];
    const unsigned ldsA1 = (unsigned)(uintptr_t)&As[1][0];
    const unsigned ldsB0 = (unsigned)(uintptr_t)&Bs[0][0];
    const unsigned ldsB1 = (unsigned)(uintptr_t)&Bs[1][0];

    v8f acc[2][4];
#pragma unroll
    for (int i = 0; i < 2; ++i)
#pragma unroll
        for (int j = 0; j < 4; ++j) acc[i][j] = (v8f){0,0,0,0,0,0,0,0};

    // prologue: prefetch K-slabs 0 and 1 (waves 0/1 drive the TDM)
    if (wave == 0) {
        tdm_load_tile(exb, rowBase, 0, Nx, ldsA0);
        tdm_load_tile(exb, rowBase, BK, Nx, ldsA1);
    }
    if (wave == 1) {
        tdm_load_tile(eyb, colBase, 0, Ny, ldsB0);
        tdm_load_tile(eyb, colBase, BK, Ny, ldsB1);
    }

    for (int kt = 0; kt < NK; ++kt) {
        const int cur = kt & 1;
        // tensor loads from one wave complete in order: <=1 outstanding => slab kt landed
        if (kt == NK - 1) __builtin_amdgcn_s_wait_tensorcnt(0);
        else              __builtin_amdgcn_s_wait_tensorcnt(1);
        __syncthreads();                         // cur buffer ready for everyone

        // two K=32 chunks per buffered slab: 16 WMMA per barrier pair
        wmma_step(&As[cur][0], &Bs[cur][0], wm, wn, m, kA0,      acc);
        wmma_step(&As[cur][0], &Bs[cur][0], wm, wn, m, kA0 + 32, acc);

        __syncthreads();                         // everyone done reading cur
        if (kt + 2 < NK) {                       // refill cur for step kt+2
            int k0 = (kt + 2) * BK;
            if (wave == 0) tdm_load_tile(exb, rowBase, k0, Nx, cur ? ldsA1 : ldsA0);
            if (wave == 1) tdm_load_tile(eyb, colBase, k0, Ny, cur ? ldsB1 : ldsB0);
        }
    }

    rowmax_epilogue(acc, rowmaxU, rowBase, wm, lane, hi);
}

// ---------------- kernel 3b: fallback GEMM (on-the-fly f32->bf16), small-ws path ----------
__global__ __launch_bounds__(256)
void gemm_fallback_kernel(const float* __restrict__ ex,
                          const float* __restrict__ ey,
                          const float* __restrict__ rnx,
                          const float* __restrict__ rny,
                          unsigned* __restrict__ rowmaxU) {
    __shared__ unsigned short As[BM * LDSS];
    __shared__ unsigned short Bs[BN * LDSS];

    const int t    = threadIdx.x;
    const int lane = t & 31;
    const int wave = t >> 5;
    const int wm   = wave >> 1;
    const int wn   = wave & 1;
    const int m    = lane & 15;
    const int hi   = lane >> 4;
    const int kA0  = hi * 8;

    const int rowBase = blockIdx.y * BM;
    const int colBase = blockIdx.x * BN;

    v8f acc[2][4];
#pragma unroll
    for (int i = 0; i < 2; ++i)
#pragma unroll
        for (int j = 0; j < 4; ++j) acc[i][j] = (v8f){0,0,0,0,0,0,0,0};

    for (int k0 = 0; k0 < KDIM; k0 += BK) {
        __syncthreads();
        // stage 128x64 slabs of ex and ey, scaled by 1/norm, as bf16
#pragma unroll
        for (int i = 0; i < 8; ++i) {
            int idx = t + i * 256;          // 0..2047
            int row = idx >> 4;             // 0..127
            int c4  = idx & 15;             // float4 column within the slab
            {
                const float4 f = *(const float4*)(ex + (size_t)(rowBase + row) * KDIM + k0 + c4 * 4);
                float s = rnx[rowBase + row];
                uint2 p; p.x = pack2bf(f.x * s, f.y * s); p.y = pack2bf(f.z * s, f.w * s);
                *(uint2*)&As[row * LDSS + c4 * 4] = p;
            }
            {
                const float4 f = *(const float4*)(ey + (size_t)(colBase + row) * KDIM + k0 + c4 * 4);
                float s = rny[colBase + row];
                uint2 p; p.x = pack2bf(f.x * s, f.y * s); p.y = pack2bf(f.z * s, f.w * s);
                *(uint2*)&Bs[row * LDSS + c4 * 4] = p;
            }
        }
        __syncthreads();
        wmma_step(As, Bs, wm, wn, m, kA0,      acc);
        wmma_step(As, Bs, wm, wn, m, kA0 + 32, acc);
    }

    rowmax_epilogue(acc, rowmaxU, rowBase, wm, lane, hi);
}

// ---------------- kernel for fallback path: reciprocal norms ----------------
__global__ void norms_kernel(const float* __restrict__ ex,
                             const float* __restrict__ ey,
                             float* __restrict__ rnx, float* __restrict__ rny,
                             int Nx, int Ny) {
    int gw   = (blockIdx.x * blockDim.x + threadIdx.x) >> 5;
    int lane = threadIdx.x & 31;
    if (gw >= Nx + Ny) return;
    const float* src = (gw < Nx) ? (ex + (size_t)gw * KDIM) : (ey + (size_t)(gw - Nx) * KDIM);
    float s = 0.f;
    const float4* src4 = (const float4*)src;
#pragma unroll
    for (int i = 0; i < 6; ++i) {
        float4 f = src4[lane + i * 32];
        s += f.x * f.x + f.y * f.y + f.z * f.z + f.w * f.w;
    }
#pragma unroll
    for (int o = 16; o > 0; o >>= 1) s += __shfl_xor(s, o, 32);
    if (lane == 0) {
        float r = 1.0f / fmaxf(sqrtf(s), 1e-8f);
        if (gw < Nx) rnx[gw] = r; else rny[gw - Nx] = r;
    }
}

// ---------------- kernel 4: entropy reduction ----------------
__global__ void finalize_kernel(const unsigned* __restrict__ rowmaxU,
                                float* __restrict__ out, int n) {
    __shared__ float red[1024];
    int t = threadIdx.x;
    float s = 0.f;
    for (int i = t; i < n; i += 1024) {
        float C = ord2f(rowmaxU[i]);
        float x = 1.0f - C;
        float l = LOGC - 0.5f * x * x;
        s += -__expf(l) * l;
    }
    red[t] = s;
    __syncthreads();
#pragma unroll
    for (int o = 512; o > 0; o >>= 1) {
        if (t < o) red[t] += red[t + o];
        __syncthreads();
    }
    if (t == 0) { out[0] = red[0]; out[1] = red[0]; }
}

// ---------------- launcher ----------------
extern "C" void kernel_launch(void* const* d_in, const int* in_sizes, int n_in,
                              void* d_out, int out_size, void* d_ws, size_t ws_size,
                              hipStream_t stream) {
    const float* ex = (const float*)d_in[0];
    const float* ey = (const float*)d_in[1];
    const int Nx = in_sizes[0] / KDIM;   // 8192
    const int Ny = in_sizes[1] / KDIM;   // 16384
    float* out = (float*)d_out;

    const size_t needFast = (size_t)(Nx + Ny) * KDIM * 2 + (size_t)Nx * 4;
    const int totalRows = Nx + Ny;
    const int nthreads  = totalRows * 32;
    dim3 grid(Ny / BN, Nx / BM);         // (128, 64)

    if (ws_size >= needFast) {
        // fast path: one-time scaled bf16 conversion + TDM-fed WMMA GEMM
        unsigned short* exb = (unsigned short*)d_ws;
        unsigned short* eyb = exb + (size_t)Nx * KDIM;
        unsigned* rowmaxU   = (unsigned*)(eyb + (size_t)Ny * KDIM);

        convert_kernel<<<(nthreads + 255) / 256, 256, 0, stream>>>(ex, ey, exb, eyb, Nx, Ny);
        init_kernel<<<(Nx + 255) / 256, 256, 0, stream>>>(rowmaxU, Nx);
        gemm_tdm_kernel<<<grid, 256, 0, stream>>>(exb, eyb, rowmaxU, Nx, Ny);
        finalize_kernel<<<1, 1024, 0, stream>>>(rowmaxU, out, Nx);
    } else {
        // small-workspace fallback: convert tiles on the fly
        float*    rnx     = (float*)d_ws;
        float*    rny     = rnx + Nx;
        unsigned* rowmaxU = (unsigned*)(rny + Ny);

        norms_kernel<<<(nthreads + 255) / 256, 256, 0, stream>>>(ex, ey, rnx, rny, Nx, Ny);
        init_kernel<<<(Nx + 255) / 256, 256, 0, stream>>>(rowmaxU, Nx);
        gemm_fallback_kernel<<<grid, 256, 0, stream>>>(ex, ey, rnx, rny, rowmaxU);
        finalize_kernel<<<1, 1024, 0, stream>>>(rowmaxU, out, Nx);
    }
}
